// DynToken_53395033424487
// MI455X (gfx1250) — compile-verified, matrix operand order checked by reading, not run on hardware
//
#include <hip/hip_runtime.h>
#include <hip/hip_bf16.h>

// ---------------------------------------------------------------------------
// Problem constants
// ---------------------------------------------------------------------------
#define BB   8
#define SA   513
#define SI   197
#define DD   768
#define HH   12
#define LL   4
#define CC   309
#define NPA  512
#define NPI  196
#define NP   708        // NPA + NPI
#define DH   64         // DD / HH
#define DFF  3072       // 4*DD

typedef __attribute__((ext_vector_type(16))) _Float16 v16h;
typedef __attribute__((ext_vector_type(8)))  float    v8f;
typedef __attribute__((ext_vector_type(2)))  __fp16   fp16x2;

// pack two f32 -> packed 2xf16 in a uint (single v_cvt_pk_rtz_f16_f32)
__device__ __forceinline__ unsigned pkh(float a, float b)
{
    union { fp16x2 h; unsigned u; } x;
    x.h = __builtin_amdgcn_cvt_pkrtz(a, b);
    return x.u;
}

// ---------------------------------------------------------------------------
// Generic WMMA GEMM:  C[z] = epi( A[z] * B[z] + bias ) (+ residual)
//   block = 256 threads = 8 wave32 waves arranged 4(M) x 2(N); each wave owns
//   a 16x32 output tile (two 16x16 accumulators sharing one A fragment) ->
//   block tile 64x64.  f16 operands, f32 accumulation via
//   v_wmma_f32_16x16x32_f16.
//   B staging:
//     transB : rows of B are contiguous in k -> f16 LDS tile [n][k]
//     !transB: rows of B are contiguous in n -> f32 LDS tile [k][n] staged
//              row-wise with ONE pointer; transpose happens at fragment
//              gather via cvt_pk_rtz from two LDS floats.
// ---------------------------------------------------------------------------
__global__ void __launch_bounds__(256)
wmma_gemm_kernel(const float* __restrict__ A, const float* __restrict__ Bm,
                 const float* __restrict__ bias, const float* __restrict__ Res,
                 float* __restrict__ C,
                 int M, int N, int K,
                 int lda, int ldb, int ldc, int ldr,
                 long long sA, long long sB, long long sC, long long sR,
                 int transB, int gelu, int hasRes)
{
    __shared__ __align__(16) _Float16 Al[8][16][32];   // A tile f16 [m][k]
    __shared__ __align__(16) float    Bt[8][32][36];   // B tile f32 [k][n+pad]
                                                       // (transB reuses as f16 [n][k])

    const int w    = threadIdx.x >> 5;
    const int lane = threadIdx.x & 31;
    const int wm   = w >> 1;           // 0..3
    const int wn   = w & 1;            // 0..1
    const int tM   = blockIdx.y * 64 + wm * 16;
    const int tN0  = blockIdx.x * 64 + wn * 32;
    const long long z = blockIdx.z;

    const float* Ab = A  + z * sA;
    const float* Bb = Bm + z * sB;
    float*       Cb = C  + z * sC;
    const float* Rb = hasRes ? (Res + z * sR) : nullptr;

    v8f acc0 = {};
    v8f acc1 = {};

    // ---- loop-invariant staging state --------------------------------------
    // A: lane covers (row rA, k cA..cA+15)
    const int  rA   = lane >> 1;
    const int  cA   = (lane & 1) * 16;
    const int  grA  = tM + rA;
    const bool rokA = (grA < M);
    const float* rowA = Ab + (long long)(rokA ? grA : (M - 1)) * lda;
    unsigned* dstA = (unsigned*)&Al[w][rA][cA];

    const bool vecA  = ((lda & 3) == 0);
    const bool vecB  = ((ldb & 3) == 0);
    const int  Kfull = K & ~31;

    const int fr  = lane & 15;
    const int kh2 = lane >> 4;

    // ---- shared staging helpers -------------------------------------------
    auto stage_A_full = [&](int k0) {
        if (vecA) {
            const float4* s4 = (const float4*)(rowA + k0 + cA);
            #pragma unroll
            for (int q = 0; q < 4; ++q) {
                float4 f = s4[q];
                if (!rokA) { f.x = 0.f; f.y = 0.f; f.z = 0.f; f.w = 0.f; }
                dstA[2 * q]     = pkh(f.x, f.y);
                dstA[2 * q + 1] = pkh(f.z, f.w);
            }
        } else {
            const float* s = rowA + k0 + cA;
            #pragma unroll
            for (int q = 0; q < 8; ++q) {
                float f0 = s[2 * q];
                float f1 = s[2 * q + 1];
                dstA[q] = pkh(rokA ? f0 : 0.0f, rokA ? f1 : 0.0f);
            }
        }
    };
    auto stage_A_tail = [&](int k0) {
        #pragma unroll
        for (int q = 0; q < 8; ++q) {
            const int c0 = k0 + cA + 2 * q;
            const int c1 = c0 + 1;
            float f0 = rowA[min(c0, K - 1)];
            float f1 = rowA[min(c1, K - 1)];
            f0 = (rokA && c0 < K) ? f0 : 0.0f;
            f1 = (rokA && c1 < K) ? f1 : 0.0f;
            dstA[q] = pkh(f0, f1);
        }
    };
    auto make_af = [&]() -> v16h {
        v16h af;
        #pragma unroll
        for (int p = 0; p < 8; ++p) {
            const int kk = (p < 4) ? (kh2 * 8 + 2 * p)
                                   : (16 + kh2 * 8 + 2 * (p - 4));
            af[2 * p]     = Al[w][fr][kk];
            af[2 * p + 1] = Al[w][fr][kk + 1];
        }
        return af;
    };

    if (transB) {
        // ================== transB: f16 LDS tile [n][k] =====================
        _Float16 (*Bh)[32] = (_Float16 (*)[32])&Bt[w][0][0];
        const int  gnT  = tN0 + lane;
        const bool nokT = (gnT < N);
        const float* rowBt = Bb + (long long)(nokT ? gnT : (N - 1)) * ldb;
        unsigned* dstBt = (unsigned*)&Bh[lane][0];

        auto mma_t = [&]() {
            const v16h af = make_af();
            v16h bf0, bf1;
            #pragma unroll
            for (int p = 0; p < 8; ++p) {
                const int kk = (p < 4) ? (kh2 * 8 + 2 * p)
                                       : (16 + kh2 * 8 + 2 * (p - 4));
                bf0[2 * p]     = Bh[fr][kk];
                bf0[2 * p + 1] = Bh[fr][kk + 1];
                bf1[2 * p]     = Bh[fr + 16][kk];
                bf1[2 * p + 1] = Bh[fr + 16][kk + 1];
            }
            acc0 = __builtin_amdgcn_wmma_f32_16x16x32_f16(
                false, af, false, bf0, (short)0, acc0, false, false);
            acc1 = __builtin_amdgcn_wmma_f32_16x16x32_f16(
                false, af, false, bf1, (short)0, acc1, false, false);
        };

        for (int k0 = 0; k0 < Kfull; k0 += 32) {
            __builtin_prefetch(rowA + k0 + 32 + cA, 0, 1);
            __builtin_prefetch(rowBt + k0 + 32, 0, 1);
            stage_A_full(k0);
            if (vecB) {
                const float4* s4 = (const float4*)(rowBt + k0);
                #pragma unroll
                for (int q = 0; q < 8; ++q) {
                    float4 f = s4[q];
                    if (!nokT) { f.x = 0.f; f.y = 0.f; f.z = 0.f; f.w = 0.f; }
                    dstBt[2 * q]     = pkh(f.x, f.y);
                    dstBt[2 * q + 1] = pkh(f.z, f.w);
                }
            } else {
                const float* s = rowBt + k0;
                #pragma unroll
                for (int q = 0; q < 16; ++q) {
                    float f0 = s[2 * q];
                    float f1 = s[2 * q + 1];
                    dstBt[q] = pkh(nokT ? f0 : 0.0f, nokT ? f1 : 0.0f);
                }
            }
            mma_t();
        }
        if (Kfull < K) {
            const int k0 = Kfull;
            stage_A_tail(k0);
            #pragma unroll
            for (int q = 0; q < 16; ++q) {
                const int c0 = k0 + 2 * q;
                const int c1 = c0 + 1;
                float f0 = rowBt[min(c0, K - 1)];
                float f1 = rowBt[min(c1, K - 1)];
                f0 = (nokT && c0 < K) ? f0 : 0.0f;
                f1 = (nokT && c1 < K) ? f1 : 0.0f;
                dstBt[q] = pkh(f0, f1);
            }
            mma_t();
        }
    } else {
        // ================== !transB: f32 LDS tile [k][n] ====================
        // lane stages row k=lane of the tile (contiguous 32 floats in memory)
        const bool fullN = (tN0 + 32 <= N);       // wave-uniform
        const float* rowBn = Bb + tN0;            // + (k0+lane)*ldb at use
        float* dstf = &Bt[w][lane][0];

        auto mma_n = [&]() {
            const v16h af = make_af();
            union { v16h h; unsigned u[8]; } b0, b1;
            #pragma unroll
            for (int p = 0; p < 8; ++p) {
                const int kk = (p < 4) ? (kh2 * 8 + 2 * p)
                                       : (16 + kh2 * 8 + 2 * (p - 4));
                b0.u[p] = pkh(Bt[w][kk][fr],      Bt[w][kk + 1][fr]);
                b1.u[p] = pkh(Bt[w][kk][fr + 16], Bt[w][kk + 1][fr + 16]);
            }
            acc0 = __builtin_amdgcn_wmma_f32_16x16x32_f16(
                false, af, false, b0.h, (short)0, acc0, false, false);
            acc1 = __builtin_amdgcn_wmma_f32_16x16x32_f16(
                false, af, false, b1.h, (short)0, acc1, false, false);
        };

        for (int k0 = 0; k0 < Kfull; k0 += 32) {
            __builtin_prefetch(rowA + k0 + 32 + cA, 0, 1);
            const float* src = rowBn + (long long)(k0 + lane) * ldb;
            __builtin_prefetch(src + (long long)32 * ldb, 0, 1);
            stage_A_full(k0);
            if (fullN && vecB) {
                const float4* s4 = (const float4*)src;
                float4* d4 = (float4*)dstf;
                #pragma unroll
                for (int q = 0; q < 8; ++q) d4[q] = s4[q];
            } else {
                #pragma unroll
                for (int j = 0; j < 32; ++j) {
                    const int gn = tN0 + j;                  // uniform
                    float f = Bb[(long long)(k0 + lane) * ldb + min(gn, N - 1)];
                    dstf[j] = (gn < N) ? f : 0.0f;
                }
            }
            mma_n();
        }
        if (Kfull < K) {
            const int k0 = Kfull;
            stage_A_tail(k0);
            const int  gk  = k0 + lane;
            const bool kok = (gk < K);
            const long long rowoff = (long long)min(gk, K - 1) * ldb;
            #pragma unroll
            for (int j = 0; j < 32; ++j) {
                const int gn = tN0 + j;                      // uniform
                float f = Bb[rowoff + min(gn, N - 1)];
                dstf[j] = (kok && gn < N) ? f : 0.0f;
            }
            mma_n();
        }
    }

    // -------- epilogue + store (C layout: VGPR v -> M = v or v+8) ----------
    const int rbase = tM + 8 * (lane >> 4);
    #pragma unroll
    for (int half = 0; half < 2; ++half) {
        const int col = tN0 + 16 * half + (lane & 15);
        const v8f acc = half ? acc1 : acc0;
        if (col < N) {
            const float bv = bias ? bias[col] : 0.0f;
            #pragma unroll
            for (int v = 0; v < 8; ++v) {
                const int row = rbase + v;
                if (row < M) {
                    float val = acc[v] + bv;
                    if (gelu) {
                        float x = val;
                        float t = tanhf(0.7978845608028654f * (x + 0.044715f * x * x * x));
                        val = 0.5f * x * (1.0f + t);
                    }
                    if (hasRes) val += Rb[(long long)row * ldr + col];
                    Cb[(long long)row * ldc + col] = val;
                }
            }
        }
    }
}

// ---------------------------------------------------------------------------
// LayerNorm: one block per row
// ---------------------------------------------------------------------------
__global__ void __launch_bounds__(256)
ln_kernel(const float* __restrict__ X, const float* __restrict__ g,
          const float* __restrict__ b, float* __restrict__ Y,
          int D, long long xstride, long long ystride)
{
    __shared__ float red[256];
    const int t = threadIdx.x;
    const float* x = X + (long long)blockIdx.x * xstride;
    float*       y = Y + (long long)blockIdx.x * ystride;

    float s = 0.0f;
    for (int i = t; i < D; i += 256) s += x[i];
    red[t] = s; __syncthreads();
    for (int o = 128; o > 0; o >>= 1) { if (t < o) red[t] += red[t + o]; __syncthreads(); }
    const float mean = red[0] / (float)D;
    __syncthreads();

    s = 0.0f;
    for (int i = t; i < D; i += 256) { float d = x[i] - mean; s += d * d; }
    red[t] = s; __syncthreads();
    for (int o = 128; o > 0; o >>= 1) { if (t < o) red[t] += red[t + o]; __syncthreads(); }
    const float rstd = rsqrtf(red[0] / (float)D + 1e-6f);

    for (int i = t; i < D; i += 256)
        y[i] = (x[i] - mean) * rstd * g[i] + b[i];
}

// ---------------------------------------------------------------------------
// Policy-masked softmax (per attention row), folds in dh^-0.5 scale.
// ---------------------------------------------------------------------------
__global__ void __launch_bounds__(256)
softmax_policy_kernel(float* __restrict__ attn, const float* __restrict__ pol,
                      int Nseq, float scale)
{
    const int q = blockIdx.x, h = blockIdx.y, t = threadIdx.x;
    float* row = attn + ((long long)h * Nseq + q) * (long long)Nseq;
    __shared__ float red[256];

    float m = -3.4e38f;
    for (int n = t; n < Nseq; n += 256) m = fmaxf(m, row[n] * scale);
    red[t] = m; __syncthreads();
    for (int o = 128; o > 0; o >>= 1) { if (t < o) red[t] = fmaxf(red[t], red[t + o]); __syncthreads(); }
    const float mx = red[0];
    __syncthreads();

    float s = 0.0f;
    for (int n = t; n < Nseq; n += 256) {
        const float p  = pol[n];
        const float ap = p + (1.0f - p) * ((n == q) ? 1.0f : 0.0f);
        const float e  = __expf(row[n] * scale - mx) * ap;
        row[n] = e; s += e;
    }
    red[t] = s; __syncthreads();
    for (int o = 128; o > 0; o >>= 1) { if (t < o) red[t] += red[t + o]; __syncthreads(); }
    const float denom = red[0] + 1e-6f;
    const float add   = 1e-6f / (float)Nseq;
    for (int n = t; n < Nseq; n += 256) row[n] = (row[n] + add) / denom;
}

// ---------------------------------------------------------------------------
// Small elementwise helpers
// ---------------------------------------------------------------------------
__global__ void fill_kernel(float* __restrict__ p, float v, int n)
{
    int i = blockIdx.x * 256 + threadIdx.x;
    if (i < n) p[i] = v;
}

__global__ void gather_spatial_kernel(const float* __restrict__ xa,
                                      const float* __restrict__ xi,
                                      float* __restrict__ out)
{
    long long i = (long long)blockIdx.x * 256 + threadIdx.x;
    const long long total = (long long)BB * NP * DD;
    if (i >= total) return;
    const int d  = (int)(i % DD);
    const long long bn = i / DD;
    const int n  = (int)(bn % NP);
    const int b  = (int)(bn / NP);
    out[i] = (n < NPA)
        ? xa[((long long)b * SA + 1 + n) * DD + d]
        : xi[((long long)b * SI + 1 + (n - NPA)) * DD + d];
}

__global__ void __launch_bounds__(256)
pool_kernel(const float* __restrict__ h, const float* __restrict__ prev,
            float* __restrict__ glob)
{
    const int d2 = blockIdx.x, b = blockIdx.y, t = threadIdx.x;
    __shared__ float r1[256], r2[256];
    float s = 0.0f, sp = 0.0f;
    for (int n = t; n < NP; n += 256) {
        const float p = prev[b * NP + n];
        s  += h[((long long)b * NP + n) * DD + (DD / 2) + d2] * p;
        sp += p;
    }
    r1[t] = s; r2[t] = sp; __syncthreads();
    for (int o = 128; o > 0; o >>= 1) {
        if (t < o) { r1[t] += r1[t + o]; r2[t] += r2[t + o]; }
        __syncthreads();
    }
    if (t == 0) glob[b * (DD / 2) + d2] = r1[0] / r2[0];
}

__global__ void hcat_kernel(const float* __restrict__ h,
                            const float* __restrict__ glob,
                            float* __restrict__ out)
{
    long long i = (long long)blockIdx.x * 256 + threadIdx.x;
    const long long total = (long long)BB * NP * DD;
    if (i >= total) return;
    const int d = (int)(i % DD);
    const int b = (int)((i / DD) / NP);
    out[i] = (d < DD / 2) ? h[i] : glob[b * (DD / 2) + (d - DD / 2)];
}

// o3 projection (192 -> 2) + gumbel argmax; forward value of the straight-
// through estimator is exactly the hard one-hot, and argmax(log_softmax+g)
// == argmax(logits+g). keep = (class0 wins) * prev.
__global__ void pred_finish_kernel(const float* __restrict__ h2,
                                   const float* __restrict__ w3,
                                   const float* __restrict__ b3,
                                   const float* __restrict__ gum,
                                   float* __restrict__ prev,
                                   float* __restrict__ preds_out,
                                   float* __restrict__ pol_a,
                                   float* __restrict__ pol_i)
{
    const int idx = blockIdx.x * 256 + threadIdx.x;
    if (idx >= BB * NP) return;
    const int b = idx / NP, n = idx % NP;
    const float* hv = h2 + (long long)idx * (DD / 4);
    float l0 = b3[0], l1 = b3[1];
    for (int k = 0; k < DD / 4; ++k) {
        const float x = hv[k];
        l0 += x * w3[k * 2 + 0];
        l1 += x * w3[k * 2 + 1];
    }
    const float g0 = gum[(long long)idx * 2 + 0];
    const float g1 = gum[(long long)idx * 2 + 1];
    const float keep = ((l0 + g0) >= (l1 + g1)) ? prev[idx] : 0.0f;
    prev[idx]      = keep;
    preds_out[idx] = keep;
    if (n < NPA) pol_a[b * SA + 1 + n]         = keep;
    else         pol_i[b * SI + 1 + (n - NPA)] = keep;
}

// ---------------------------------------------------------------------------
// Host orchestration
// ---------------------------------------------------------------------------
static inline void gemm(hipStream_t s,
                        const float* A, const float* B, const float* bias,
                        const float* Res, float* C,
                        int M, int N, int K,
                        int lda, int ldb, int ldc, int ldr,
                        long long sA, long long sB, long long sC, long long sR,
                        int batch, int transB, int gelu)
{
    dim3 grid((N + 63) / 64, (M + 63) / 64, batch);
    wmma_gemm_kernel<<<grid, 256, 0, s>>>(A, B, bias, Res, C,
                                          M, N, K, lda, ldb, ldc, ldr,
                                          sA, sB, sC, sR,
                                          transB, gelu, Res != nullptr);
}

extern "C" void kernel_launch(void* const* d_in, const int* in_sizes, int n_in,
                              void* d_out, int out_size, void* d_ws, size_t ws_size,
                              hipStream_t stream)
{
    (void)in_sizes; (void)n_in; (void)out_size; (void)ws_size;

    // input pointer map (dict insertion order, nested dicts flattened)
    const float* audio  = (const float*)d_in[0];
    const float* image  = (const float*)d_in[1];
    const float* gumbel = (const float*)d_in[2];
    const int IA = 3;    // pa: ln1_g,ln1_b,qkv_w,qkv_b,proj_w,proj_b,ln2_g,ln2_b,fc1_w,fc1_b,fc2_w,fc2_b
    const int II = 15;   // pi: same layout
    const int IP = 27;   // pp: ln_g,ln_b,in_w,in_b,o1_w,o1_b,o2_w,o2_b,o3_w,o3_b
    const int IH = 37;   // ph: norm_a_g,norm_a_b,norm_i_g,norm_i_b,head_w,head_b
    float* out = (float*)d_out;

    // workspace bump allocator (floats)
    float* w = (float*)d_ws;
    size_t off = 0;
    auto alloc = [&](size_t n) { float* p = w + off; off += n; return p; };
    float* xa   = alloc((size_t)BB * SA * DD);
    float* xi   = alloc((size_t)BB * SI * DD);
    float* hbuf = alloc((size_t)BB * SA * DD);
    float* qkv  = alloc((size_t)BB * SA * 3 * DD);
    float* attn = alloc((size_t)HH * SA * SA);
    float* obuf = alloc((size_t)BB * SA * DD);
    float* mlp  = alloc((size_t)BB * SA * DFF);
    float* hA   = alloc((size_t)BB * NP * DD);
    float* hB   = alloc((size_t)BB * NP * DD);
    float* hp1  = alloc((size_t)BB * NP * (DD / 2));
    float* hp2  = alloc((size_t)BB * NP * (DD / 4));
    float* glob = alloc((size_t)BB * (DD / 2));
    float* prev = alloc((size_t)BB * NP);
    float* pola = alloc((size_t)BB * SA);
    float* poli = alloc((size_t)BB * SI);
    float* clsb = alloc((size_t)BB * 2 * DD);

    // init activations & policies
    (void)hipMemcpyAsync(xa, audio, sizeof(float) * (size_t)BB * SA * DD, hipMemcpyDeviceToDevice, stream);
    (void)hipMemcpyAsync(xi, image, sizeof(float) * (size_t)BB * SI * DD, hipMemcpyDeviceToDevice, stream);
    fill_kernel<<<(BB * NP + 255) / 256, 256, 0, stream>>>(prev, 1.0f, BB * NP);
    fill_kernel<<<(BB * SA + 255) / 256, 256, 0, stream>>>(pola, 1.0f, BB * SA);
    fill_kernel<<<(BB * SI + 255) / 256, 256, 0, stream>>>(poli, 1.0f, BB * SI);

    // transformer block for one tower
    auto run_block = [&](float* x, int Nseq, int base, int l, const float* pol) {
        const int rows = BB * Nseq;
        const float* ln1g = (const float*)d_in[base + 0] + (size_t)l * DD;
        const float* ln1b = (const float*)d_in[base + 1] + (size_t)l * DD;
        const float* qkvw = (const float*)d_in[base + 2] + (size_t)l * DD * 3 * DD;
        const float* qkvb = (const float*)d_in[base + 3] + (size_t)l * 3 * DD;
        const float* prjw = (const float*)d_in[base + 4] + (size_t)l * DD * DD;
        const float* prjb = (const float*)d_in[base + 5] + (size_t)l * DD;
        const float* ln2g = (const float*)d_in[base + 6] + (size_t)l * DD;
        const float* ln2b = (const float*)d_in[base + 7] + (size_t)l * DD;
        const float* fc1w = (const float*)d_in[base + 8] + (size_t)l * DD * DFF;
        const float* fc1b = (const float*)d_in[base + 9] + (size_t)l * DFF;
        const float* fc2w = (const float*)d_in[base + 10] + (size_t)l * DFF * DD;
        const float* fc2b = (const float*)d_in[base + 11] + (size_t)l * DD;

        // h = LN1(x)
        ln_kernel<<<rows, 256, 0, stream>>>(x, ln1g, ln1b, hbuf, DD, DD, DD);
        // qkv = h @ qkv_w + qkv_b
        gemm(stream, hbuf, qkvw, qkvb, nullptr, qkv,
             rows, 3 * DD, DD, DD, 3 * DD, 3 * DD, 0, 0, 0, 0, 0, 1, 0, 0);
        // attention, looped over batch b (heads via grid.z)
        for (int b = 0; b < BB; ++b) {
            const float* qb = qkv + (size_t)b * Nseq * 3 * DD;
            // scores[h,q,k'] = sum_d q . k   (B accessed transposed)
            gemm(stream, qb, qb + DD, nullptr, nullptr, attn,
                 Nseq, Nseq, DH, 3 * DD, 3 * DD, Nseq, 0,
                 DH, DH, (long long)Nseq * Nseq, 0, HH, 1, 0);
            softmax_policy_kernel<<<dim3(Nseq, HH), 256, 0, stream>>>(
                attn, pol + (size_t)b * Nseq, Nseq, 0.125f);
            // o[b,q,h*64+d] = attn @ v
            gemm(stream, attn, qb + 2 * DD, nullptr, nullptr,
                 obuf + (size_t)b * Nseq * DD,
                 Nseq, DH, Nseq, Nseq, 3 * DD, DD, 0,
                 (long long)Nseq * Nseq, DH, DH, 0, HH, 0, 0);
        }
        // x = x + o @ proj_w + proj_b
        gemm(stream, obuf, prjw, prjb, x, x,
             rows, DD, DD, DD, DD, DD, DD, 0, 0, 0, 0, 1, 0, 0);
        // h = LN2(x)
        ln_kernel<<<rows, 256, 0, stream>>>(x, ln2g, ln2b, hbuf, DD, DD, DD);
        // mlp = gelu(h @ fc1 + b1)
        gemm(stream, hbuf, fc1w, fc1b, nullptr, mlp,
             rows, DFF, DD, DD, DFF, DFF, 0, 0, 0, 0, 0, 1, 0, 1);
        // x = x + mlp @ fc2 + b2
        gemm(stream, mlp, fc2w, fc2b, x, x,
             rows, DD, DFF, DFF, DD, DD, DD, 0, 0, 0, 0, 1, 0, 0);
    };

    // token predictor at PRUNE layers
    auto run_predictor = [&](int j) {
        const float* lng = (const float*)d_in[IP + 0] + (size_t)j * DD;
        const float* lnb = (const float*)d_in[IP + 1] + (size_t)j * DD;
        const float* inw = (const float*)d_in[IP + 2] + (size_t)j * DD * DD;
        const float* inb = (const float*)d_in[IP + 3] + (size_t)j * DD;
        const float* o1w = (const float*)d_in[IP + 4] + (size_t)j * DD * (DD / 2);
        const float* o1b = (const float*)d_in[IP + 5] + (size_t)j * (DD / 2);
        const float* o2w = (const float*)d_in[IP + 6] + (size_t)j * (DD / 2) * (DD / 4);
        const float* o2b = (const float*)d_in[IP + 7] + (size_t)j * (DD / 4);
        const float* o3w = (const float*)d_in[IP + 8] + (size_t)j * (DD / 4) * 2;
        const float* o3b = (const float*)d_in[IP + 9] + (size_t)j * 2;
        const int rows = BB * NP;
        const long long total = (long long)rows * DD;

        gather_spatial_kernel<<<(int)((total + 255) / 256), 256, 0, stream>>>(xa, xi, hA);
        ln_kernel<<<rows, 256, 0, stream>>>(hA, lng, lnb, hB, DD, DD, DD);
        gemm(stream, hB, inw, inb, nullptr, hA,
             rows, DD, DD, DD, DD, DD, 0, 0, 0, 0, 0, 1, 0, 1);      // gelu
        pool_kernel<<<dim3(DD / 2, BB), 256, 0, stream>>>(hA, prev, glob);
        hcat_kernel<<<(int)((total + 255) / 256), 256, 0, stream>>>(hA, glob, hB);
        gemm(stream, hB, o1w, o1b, nullptr, hp1,
             rows, DD / 2, DD, DD, DD / 2, DD / 2, 0, 0, 0, 0, 0, 1, 0, 1);
        gemm(stream, hp1, o2w, o2b, nullptr, hp2,
             rows, DD / 4, DD / 2, DD / 2, DD / 4, DD / 4, 0, 0, 0, 0, 0, 1, 0, 1);
        pred_finish_kernel<<<(rows + 255) / 256, 256, 0, stream>>>(
            hp2, o3w, o3b, gumbel + (size_t)j * rows * 2,
            prev, out + BB * CC + (size_t)j * rows, pola, poli);
    };

    int j = 0;
    for (int l = 0; l < LL; ++l) {
        if (l == 1 || l == 3) { run_predictor(j); ++j; }
        run_block(xa, SA, IA, l, pola);
        run_block(xi, SI, II, l, poli);
    }

    // final LN on cls tokens directly into concatenated cls buffer
    ln_kernel<<<BB, 256, 0, stream>>>(xa, (const float*)d_in[IH + 0],
                                      (const float*)d_in[IH + 1], clsb,
                                      DD, (long long)SA * DD, 2 * DD);
    ln_kernel<<<BB, 256, 0, stream>>>(xi, (const float*)d_in[IH + 2],
                                      (const float*)d_in[IH + 3], clsb + DD,
                                      DD, (long long)SI * DD, 2 * DD);
    // logits = cls @ head_w + head_b
    gemm(stream, clsb, (const float*)d_in[IH + 4], (const float*)d_in[IH + 5],
         nullptr, out, BB, CC, 2 * DD, 2 * DD, CC, CC, 0, 0, 0, 0, 0, 1, 0, 0);
}
